// SelectiveGroupModel_17386027614642
// MI455X (gfx1250) — matrix operand order repaired and checked by reading, not compiled, not run on hardware
//
#include <hip/hip_runtime.h>
#include <hip/hip_bf16.h>
#include <stdint.h>

// ---------------------------------------------------------------------------
// Problem constants (match reference)
// ---------------------------------------------------------------------------
constexpr int   Bn  = 32;
constexpr int   Tn  = 2048;
constexpr int   Wd  = 1024;
constexpr int   Gn  = 16;
constexpr int   Dn  = 64;
constexpr float MOM = 0.9f;
constexpr float TH  = 0.5f;
constexpr float EPS = 1e-8f;

constexpr size_t XBF_BYTES = (size_t)Bn * Wd * Tn * 2;   // 128 MB bf16 transposed copy
constexpr int    OUT_ELEMS = Bn * Tn * Gn;               // 1,048,576
constexpr int    CORR_OFF  = OUT_ELEMS;                  // correlations (16)
constexpr int    EMA_OFF   = OUT_ELEMS + Gn;             // ema_new (1024*1024)

#define DEVINL __device__ __forceinline__

typedef __attribute__((ext_vector_type(16))) __bf16   v16bf;
typedef __attribute__((ext_vector_type(8)))  float    v8f;
typedef __attribute__((ext_vector_type(8)))  uint32_t v8u;
typedef __attribute__((ext_vector_type(4)))  uint32_t vu4;
typedef __attribute__((ext_vector_type(8)))  int      vi8;
typedef __attribute__((ext_vector_type(4)))  int      vi4;

// ---------------------------------------------------------------------------
// bf16 / math helpers
// ---------------------------------------------------------------------------
DEVINL unsigned short f2bf(float f) {
  uint32_t u = __builtin_bit_cast(uint32_t, f);
  u += 0x7FFFu + ((u >> 16) & 1u);
  return (unsigned short)(u >> 16);
}
DEVINL float bf2f(unsigned short h) {
  uint32_t u = ((uint32_t)h) << 16;
  return __builtin_bit_cast(float, u);
}
DEVINL v16bf pack16(const unsigned short* t) {
  v8u p;
  #pragma unroll
  for (int q = 0; q < 8; ++q)
    p[q] = (uint32_t)t[2 * q] | ((uint32_t)t[2 * q + 1] << 16);
  return __builtin_bit_cast(v16bf, p);
}
// Fast reciprocal: corr is clamped to [-1,1] and thresholded at 0.5, so the
// ~1ulp v_rcp_f32 is plenty; avoids the 10-op IEEE divide expansion that was
// polluting the WMMA stream in the previous build.
DEVINL float fast_rcp(float d) {
#if __has_builtin(__builtin_amdgcn_rcpf)
  return __builtin_amdgcn_rcpf(d);
#else
  return 1.0f / d;
#endif
}

DEVINL v8f wmma_bf16(v16bf a, v16bf b, v8f c) {
  // D = A(16x32 bf16) x B(32x16 bf16) + C(16x16 f32)
  return __builtin_amdgcn_wmma_f32_16x16x32_bf16(false, a, false, b,
                                                 (short)0, c, false, false);
}

// WMMA fragment loads from LDS tiles laid out [column][t] (K contiguous,
// row length 64 bf16). kc selects the 32-wide K-chunk within the stage.
// A 16-bit A-matrix layout: lanes 0-15 (M=lane) hold K={0..7,16..23},
// lanes 16-31 (M=lane-16) hold K={8..15,24..31}  -> two 16B reads per lane.
DEVINL v16bf frag_a(unsigned short (*blk)[64], int rowbase, int kc, int lane) {
  int r  = rowbase + (lane & 15);
  int c0 = kc * 32 + ((lane & 16) ? 8 : 0);
  vu4 lo = *(const vu4*)&blk[r][c0];
  vu4 hi = *(const vu4*)&blk[r][c0 + 16];
  v8u p = { lo[0], lo[1], lo[2], lo[3], hi[0], hi[1], hi[2], hi[3] };
  return __builtin_bit_cast(v16bf, p);
}
// B 32x16 layout: lanes 0-15 (N=lane) hold K={0..7,16..23}, lanes 16-31 K={8..15,24..31}.
DEVINL v16bf frag_b(unsigned short (*blk)[64], int rowbase, int kc, int lane) {
  int r  = rowbase + (lane & 15);
  int c0 = kc * 32 + ((lane & 16) ? 8 : 0);
  vu4 lo = *(const vu4*)&blk[r][c0];
  vu4 hi = *(const vu4*)&blk[r][c0 + 16];
  v8u p = { lo[0], lo[1], lo[2], lo[3], hi[0], hi[1], hi[2], hi[3] };
  return __builtin_bit_cast(v16bf, p);
}

// ---------------------------------------------------------------------------
// Tensor Data Mover staging (gfx1250 TDM) with cooperative-load fallback
// ---------------------------------------------------------------------------
#if __has_builtin(__builtin_amdgcn_tensor_load_to_lds) && \
    __has_builtin(__builtin_amdgcn_s_wait_tensorcnt)
#define USE_TDM 1
#else
#define USE_TDM 0
#endif

#if USE_TDM
// Load a 2D tile (64 rows x 64 bf16, row stride = Tn elements) from the
// bf16 (B,W,T) tensor into LDS at lds_off. D# fields per CDNA5 ISA.
DEVINL void tdm_load_tile(uint32_t lds_off, const unsigned short* gptr) {
  unsigned long long ga = (unsigned long long)(size_t)gptr;
  vu4 g0;
  g0[0] = 1u;                                    // count=1, user descriptor
  g0[1] = lds_off;                               // lds_addr (bytes)
  g0[2] = (uint32_t)ga;                          // global_addr[31:0]
  g0[3] = ((uint32_t)(ga >> 32) & 0x01FFFFFFu) | (2u << 30);  // [56:32] | type=2
  vi8 g1;
  g1[0] = (1 << 16);                             // data_size=2B, mask=0, no pad
  g1[1] = (int)((uint32_t)(Tn & 0xFFFF) << 16);  // tensor_dim0 = Tn (lo16)
  g1[2] = (int)(((uint32_t)(Wd & 0xFFFF) << 16) | (uint32_t)(Tn >> 16)); // dim1|dim0hi
  g1[3] = (int)(((uint32_t)64 << 16) | (uint32_t)(Wd >> 16));            // tile_dim0=64
  g1[4] = 64;                                    // tile_dim1=64, tile_dim2=0
  g1[5] = Tn;                                    // tensor_dim0_stride (lo32)
  g1[6] = 0;
  g1[7] = 0;
  vi4 z4 = { 0, 0, 0, 0 };
#if __clang_major__ >= 23
  vi8 z8 = { 0, 0, 0, 0, 0, 0, 0, 0 };
  __builtin_amdgcn_tensor_load_to_lds(g0, g1, z4, z4, z8, 0);
#else
  __builtin_amdgcn_tensor_load_to_lds(g0, g1, z4, z4, 0);
#endif
}
#endif

// ---------------------------------------------------------------------------
// K0: zero the group-correlation accumulator
// ---------------------------------------------------------------------------
__global__ void k_init(float* __restrict__ corr_acc) {
  if (threadIdx.x < Gn) corr_acc[threadIdx.x] = 0.0f;
}

// ---------------------------------------------------------------------------
// K1: transpose+convert x (B,T,W) f32  ->  xbf (B,W,T) bf16 via LDS tiles
// ---------------------------------------------------------------------------
__global__ void __launch_bounds__(256) k_transpose(
    const float* __restrict__ x, unsigned short* __restrict__ xbf) {
  __shared__ unsigned short tile[64][65];  // pad to dodge bank conflicts
  const int b  = blockIdx.z;
  const int t0 = blockIdx.y * 64;
  const int w0 = blockIdx.x * 64;
  const int tid = threadIdx.x;
  #pragma unroll
  for (int k = 0; k < 16; ++k) {
    int idx = tid + k * 256;
    int tr = idx >> 6, tc = idx & 63;
    const float* p = &x[((size_t)b * Tn + t0 + tr) * Wd + w0 + tc];
    __builtin_prefetch(p + 4 * Wd, 0, 1);        // global_prefetch_b8
    tile[tr][tc] = f2bf(*p);
  }
  __syncthreads();
  #pragma unroll
  for (int k = 0; k < 16; ++k) {
    int idx = tid + k * 256;
    int wr = idx >> 6, wc = idx & 63;
    xbf[((size_t)b * Wd + w0 + wr) * Tn + t0 + wc] = tile[wc][wr];
  }
}

// ---------------------------------------------------------------------------
// K2: per-(b,w) mean / std over T from the transposed bf16 copy
// ---------------------------------------------------------------------------
__global__ void __launch_bounds__(256) k_stats(
    const unsigned short* __restrict__ xbf,
    float* __restrict__ meanv, float* __restrict__ stdv) {
  const int lane = threadIdx.x & 31;
  const int row  = blockIdx.x * 8 + (threadIdx.x >> 5);  // (b*Wd + w)
  const uint32_t* p = (const uint32_t*)(xbf + (size_t)row * Tn);
  float s = 0.0f, sq = 0.0f;
  for (int k = lane; k < Tn / 2; k += 32) {
    uint32_t u = p[k];
    float f0 = bf2f((unsigned short)(u & 0xFFFFu));
    float f1 = bf2f((unsigned short)(u >> 16));
    s += f0 + f1;
    sq += f0 * f0 + f1 * f1;
  }
  for (int m = 16; m; m >>= 1) { s += __shfl_xor(s, m); sq += __shfl_xor(sq, m); }
  if (lane == 0) {
    float var = (sq - s * s / (float)Tn) / (float)(Tn - 1);
    meanv[row] = s / (float)Tn;
    stdv[row]  = sqrtf(fmaxf(var, 0.0f));
  }
}

// ---------------------------------------------------------------------------
// K3: 64x64-tile SYRK over T per batch (WMMA bf16), TDM double-buffered:
//     wave0 issues the NEXT stage's tiles, then waits tensorcnt<=2 so only
//     the CURRENT stage's pair is forced complete (TDM completes in order),
//     overlapping DMA with WMMA on the other LDS buffer. Fused epilogue:
//     centered-cov -> clipped Pearson corr -> sequential EMA scan over b
//     (ema tile in registers) + group |off-diag corr| stat on diagonal tiles.
// Block = 128 threads (4 waves), wave grid 2x2, each wave 32x32 (2x2 frags),
// K = 64 per stage (2 WMMA chunks) to halve barrier frequency.
// ---------------------------------------------------------------------------
__global__ void __launch_bounds__(128) k_syrk_ema(
    const unsigned short* __restrict__ xbf,
    const float* __restrict__ meanv, const float* __restrict__ stdv,
    const float* __restrict__ ema_in, float* __restrict__ ema_out,
    float* __restrict__ corr_acc) {
  __shared__ unsigned short Abuf[2][64][64];   // ping-pong A tiles (8 KB each)
  __shared__ unsigned short Bbuf[2][64][64];   // ping-pong B tiles
  __shared__ float mI[64], sI[64], mJ[64], sJ[64];
  __shared__ float red[4];

  const int tid  = threadIdx.x;
  const int lane = tid & 31;
  const int wave = tid >> 5;
  const int rb = (wave >> 1) * 32;   // wave row base within 64x64 tile
  const int cb = (wave & 1) * 32;    // wave col base
  const int i0 = blockIdx.y * 64;
  const int j0 = blockIdx.x * 64;
  const bool diag = (i0 == j0);
  const int hb = (lane & 16) ? 8 : 0;  // C-layout: M = r + 8*(lane>=16)
  const int ln = lane & 15;            // C-layout: N = lane%16
  const float invTm1 = 1.0f / (float)(Tn - 1);
  const v8f vzero = { 0.f, 0.f, 0.f, 0.f, 0.f, 0.f, 0.f, 0.f };
  constexpr int NSTAGE = Tn / 64;      // 32 stages of K=64

  // EMA tile lives in registers across the whole batch scan.
  v8f ema[2][2];
  #pragma unroll
  for (int mt = 0; mt < 2; ++mt)
    #pragma unroll
    for (int nt = 0; nt < 2; ++nt)
      #pragma unroll
      for (int r = 0; r < 8; ++r) {
        int row = i0 + rb + mt * 16 + r + hb;
        int col = j0 + cb + nt * 16 + ln;
        ema[mt][nt][r] = ema_in[(size_t)row * Wd + col];
      }

  float gacc = 0.0f;

  for (int b = 0; b < Bn; ++b) {
    __syncthreads();  // previous epilogue done before restaging stats
    if (tid < 64) {
      mI[tid] = meanv[b * Wd + i0 + tid];
      sI[tid] = stdv [b * Wd + i0 + tid];
      mJ[tid] = meanv[b * Wd + j0 + tid];
      sJ[tid] = stdv [b * Wd + j0 + tid];
    }
    v8f acc[2][2];
    acc[0][0] = vzero; acc[0][1] = vzero; acc[1][0] = vzero; acc[1][1] = vzero;

    const unsigned short* Asrc = xbf + ((size_t)b * Wd + i0) * Tn;
    const unsigned short* Bsrc = xbf + ((size_t)b * Wd + j0) * Tn;

#if USE_TDM
    if (tid < 32) {   // prologue: stage 0 into buffer 0
      tdm_load_tile((uint32_t)(size_t)&Abuf[0][0][0], Asrc);
      tdm_load_tile((uint32_t)(size_t)&Bbuf[0][0][0], Bsrc);
    }
#endif
    for (int it = 0; it < NSTAGE; ++it) {
      const int cur = it & 1;
      __syncthreads();   // reads of buf[cur] from 2 stages ago are done
#if USE_TDM
      if (tid < 32) {    // wave0 drives the Tensor Data Mover pipeline
        if (it + 1 < NSTAGE) {
          tdm_load_tile((uint32_t)(size_t)&Abuf[cur ^ 1][0][0],
                        Asrc + (size_t)(it + 1) * 64);
          tdm_load_tile((uint32_t)(size_t)&Bbuf[cur ^ 1][0][0],
                        Bsrc + (size_t)(it + 1) * 64);
          __builtin_amdgcn_s_wait_tensorcnt(2);  // current pair done, next in flight
        } else {
          __builtin_amdgcn_s_wait_tensorcnt(0);
        }
      }
#else
      #pragma unroll
      for (int k = 0; k < 16; ++k) {
        int idx = tid + k * 128;          // 2048 dwords per tile
        int rr = idx >> 5, cc = idx & 31;
        ((uint32_t*)Abuf[cur])[idx] =
            ((const uint32_t*)(Asrc + (size_t)rr * Tn + it * 64))[cc];
        ((uint32_t*)Bbuf[cur])[idx] =
            ((const uint32_t*)(Bsrc + (size_t)rr * Tn + it * 64))[cc];
      }
#endif
      __syncthreads();   // buf[cur] visible to all waves
      #pragma unroll
      for (int kc = 0; kc < 2; ++kc) {
        v16bf a0 = frag_a(Abuf[cur], rb + 0,  kc, lane);
        v16bf a1 = frag_a(Abuf[cur], rb + 16, kc, lane);
        v16bf b0 = frag_b(Bbuf[cur], cb + 0,  kc, lane);
        v16bf b1 = frag_b(Bbuf[cur], cb + 16, kc, lane);
        acc[0][0] = wmma_bf16(a0, b0, acc[0][0]);
        acc[0][1] = wmma_bf16(a0, b1, acc[0][1]);
        acc[1][0] = wmma_bf16(a1, b0, acc[1][0]);
        acc[1][1] = wmma_bf16(a1, b1, acc[1][1]);
      }
    }

    // Epilogue for this batch: cov -> corr -> EMA step (+ group stat)
    #pragma unroll
    for (int mt = 0; mt < 2; ++mt)
      #pragma unroll
      for (int nt = 0; nt < 2; ++nt)
        #pragma unroll
        for (int r = 0; r < 8; ++r) {
          int rl = rb + mt * 16 + r + hb;
          int cl = cb + nt * 16 + ln;
          float cov  = (acc[mt][nt][r] - (float)Tn * mI[rl] * mJ[cl]) * invTm1;
          float corr = cov * fast_rcp(sI[rl] * sJ[cl] + EPS);
          corr = fminf(1.0f, fmaxf(-1.0f, corr));
          if (fabsf(corr) > TH)
            ema[mt][nt][r] = MOM * ema[mt][nt][r] + (1.0f - MOM) * corr;
          if (diag && (rl != cl)) gacc += fabsf(corr);
        }
  }

  // Write ema_new
  #pragma unroll
  for (int mt = 0; mt < 2; ++mt)
    #pragma unroll
    for (int nt = 0; nt < 2; ++nt)
      #pragma unroll
      for (int r = 0; r < 8; ++r) {
        int row = i0 + rb + mt * 16 + r + hb;
        int col = j0 + cb + nt * 16 + ln;
        ema_out[(size_t)row * Wd + col] = ema[mt][nt][r];
      }

  if (diag) {
    for (int m = 16; m; m >>= 1) gacc += __shfl_xor(gacc, m);
    if (lane == 0) red[wave] = gacc;
    __syncthreads();
    if (tid == 0)
      atomicAdd(&corr_acc[i0 >> 6], red[0] + red[1] + red[2] + red[3]);
  }
}

// ---------------------------------------------------------------------------
// K4: per-group expert MLP (D=64 -> 16 -> 1) via two chained WMMAs (K=64),
//     relu + W2 contraction with wave32 shfl_xor reduction.
// Block = 256 (8 waves); wave handles one 16-row tile of flat (b*T+t) rows.
// ---------------------------------------------------------------------------
__global__ void __launch_bounds__(256) k_mlp(
    const float* __restrict__ x, const float* __restrict__ W1,
    const float* __restrict__ b1, const float* __restrict__ W2,
    const float* __restrict__ b2, float* __restrict__ out) {
  const int tid  = threadIdx.x;
  const int lane = tid & 31;
  const int wave = tid >> 5;
  const int g    = blockIdx.y;
  const int tile = blockIdx.x * 8 + wave;  // 16-row tiles of the (B*T) axis
  const int ln   = lane & 15;
  const int hi16 = (lane & 16) ? 1 : 0;
  const v8f vzero = { 0.f, 0.f, 0.f, 0.f, 0.f, 0.f, 0.f, 0.f };

  // B fragments: B[k=d][n=h] = W1[g][h][d]
  const float* w1r = W1 + ((size_t)g * 16 + ln) * Dn;
  v16bf bf[2];
  #pragma unroll
  for (int f = 0; f < 2; ++f) {
    int kb = f * 32 + hi16 * 8;
    unsigned short t16[16];
    #pragma unroll
    for (int q = 0; q < 8; ++q) t16[q]     = f2bf(w1r[kb + q]);
    #pragma unroll
    for (int q = 0; q < 8; ++q) t16[8 + q] = f2bf(w1r[kb + 16 + q]);
    bf[f] = pack16(t16);
  }

  // A fragments: A[m=t][k=d] = x[row][g*64 + d]
  const int R = tile * 16 + ln;                    // flat (b*T + t)
  const float* xr = x + (size_t)R * Wd + g * Dn;
  v16bf af[2];
  #pragma unroll
  for (int f = 0; f < 2; ++f) {
    int c0 = f * 32 + hi16 * 8;
    unsigned short t16[16];
    #pragma unroll
    for (int q = 0; q < 8; ++q) t16[q]     = f2bf(xr[c0 + q]);
    #pragma unroll
    for (int q = 0; q < 8; ++q) t16[8 + q] = f2bf(xr[c0 + 16 + q]);
    af[f] = pack16(t16);
  }

  v8f c = vzero;
  c = wmma_bf16(af[0], bf[0], c);
  c = wmma_bf16(af[1], bf[1], c);

  const float b1v = b1[g * 16 + ln];
  const float w2v = W2[g * 16 + ln];
  const float b2v = b2[g];

  #pragma unroll
  for (int r = 0; r < 8; ++r) {
    float hv = fmaxf(c[r] + b1v, 0.0f) * w2v;     // relu(h)*W2[h]
    hv += __shfl_xor(hv, 1);                       // reduce over h (16 lanes;
    hv += __shfl_xor(hv, 2);                       // xor<16 stays in each half,
    hv += __shfl_xor(hv, 4);                       // matching M = r + 8*half)
    hv += __shfl_xor(hv, 8);
    if (ln == 0) {
      int M = tile * 16 + r + hi16 * 8;
      out[(size_t)M * Gn + g] = hv + b2v;
    }
  }
}

// ---------------------------------------------------------------------------
// K5: finalize correlations
// ---------------------------------------------------------------------------
__global__ void k_final(const float* __restrict__ acc, float* __restrict__ corr) {
  int g = threadIdx.x;
  if (g < Gn)
    corr[g] = acc[g] * (1.0f / ((float)(Dn * Dn - Dn) * (float)Bn));
}

// ---------------------------------------------------------------------------
// Host launcher
// ---------------------------------------------------------------------------
extern "C" void kernel_launch(void* const* d_in, const int* in_sizes, int n_in,
                              void* d_out, int out_size, void* d_ws, size_t ws_size,
                              hipStream_t stream) {
  const float* x   = (const float*)d_in[0];   // (B,T,W)
  const float* ema = (const float*)d_in[1];   // (W,W)
  const float* W1  = (const float*)d_in[2];   // (G,16,D)
  const float* b1  = (const float*)d_in[3];   // (G,16)
  const float* W2  = (const float*)d_in[4];   // (G,1,16)
  const float* b2  = (const float*)d_in[5];   // (G,1)

  float* out      = (float*)d_out;            // (B,T,G)
  float* corr_out = out + CORR_OFF;           // (G,)
  float* ema_out  = out + EMA_OFF;            // (W,W)

  // Workspace: bf16 transposed x copy + stats + group-corr accumulator
  unsigned short* xbf = (unsigned short*)d_ws;
  float* meanv = (float*)((char*)d_ws + XBF_BYTES);
  float* stdv  = meanv + (size_t)Bn * Wd;
  float* cacc  = stdv  + (size_t)Bn * Wd;

  k_init<<<1, 32, 0, stream>>>(cacc);
  k_transpose<<<dim3(Wd / 64, Tn / 64, Bn), 256, 0, stream>>>(x, xbf);
  k_stats<<<(Bn * Wd) / 8, 256, 0, stream>>>(xbf, meanv, stdv);
  k_syrk_ema<<<dim3(Wd / 64, Wd / 64), 128, 0, stream>>>(
      xbf, meanv, stdv, ema, ema_out, cacc);
  k_mlp<<<dim3((Bn * Tn / 16) / 8, Gn), 256, 0, stream>>>(
      x, W1, b1, W2, b2, out);
  k_final<<<1, 32, 0, stream>>>(cacc, corr_out);
}
